// MultiScaleCrossAttention_20925080666670
// MI455X (gfx1250) — matrix operand order repaired and checked by reading, not compiled, not run on hardware
//
#include <hip/hip_runtime.h>
#include <hip/hip_bf16.h>

// ---------------------------------------------------------------------------
// Problem constants (from reference)
// ---------------------------------------------------------------------------
#define B_SZ   2
#define NQ     8192
#define NKV    1024
#define CQ     384
#define CKV    768
#define NH     8
#define HD     48          // head dim
#define HDP    64          // head dim padded to multiple of 32 for WMMA K
#define QTILES (NQ / 16)   // 512 q-tiles of 16 rows per (b, head)
#define SCALE  0.14433756729740643f   // 48^-0.5
#define LOG2E  1.4426950408889634f

typedef _Float16 half_t;
typedef __attribute__((ext_vector_type(16))) _Float16 v16h;
typedef __attribute__((ext_vector_type(8)))  _Float16 v8h;
typedef __attribute__((ext_vector_type(8)))  float    v8f;

// ---------------------------------------------------------------------------
// WMMA fragment loaders (wave32; A/B use this code's contiguous k order, not the ISA table order)
//
// A (16x32 f16, MxK): lane L -> row m = L&15, half h = L>>4.
//   elements e=0..15 : K = k0 + 16h + e         (two contiguous 16B loads)
// B (32x16 f16, KxN): lane L -> col n = L&15, half h = L>>4.
//   elements e=0..15 : K = k0 + 16h + e         (contiguous 32B)
//   -> with B stored transposed BT[N][K], a B fragment is one contiguous load.
// C/D (16x16 f32): lane L -> col n = L&15; reg r -> row m = r + 8*(L>>4).
// ---------------------------------------------------------------------------
__device__ __forceinline__ v16h load_fragA(const half_t* row, int k0, int hh) {
    union { v16h v; v8h h[2]; } u;
    u.h[0] = *(const v8h*)(row + k0 + 16 * hh);
    u.h[1] = *(const v8h*)(row + k0 + 16 * hh + 8);
    return u.v;
}

__device__ __forceinline__ v16h load_fragB(const half_t* bt_row, int k0, int hh) {
    return *(const v16h*)(bt_row + k0 + 16 * hh);
}

__device__ __forceinline__ v8f wmma_f16(v16h a, v16h b, v8f c) {
    return __builtin_amdgcn_wmma_f32_16x16x32_f16(false, a, false, b, (short)0, c,
                                                  false, false);
}

__device__ __forceinline__ v8f vzero8() {
    v8f z = {0.f, 0.f, 0.f, 0.f, 0.f, 0.f, 0.f, 0.f};
    return z;
}

// ---------------------------------------------------------------------------
// Weight convert + transpose: W[K][N] f32  ->  WT[N][K] f16
// ---------------------------------------------------------------------------
__global__ void wconv_kernel(const float* __restrict__ W, half_t* __restrict__ WT,
                             int K, int N) {
    int idx = blockIdx.x * blockDim.x + threadIdx.x;
    if (idx < N * K) {
        int nn = idx / K;
        int kk = idx - nn * K;
        WT[idx] = (half_t)W[kk * N + nn];
    }
}

// ---------------------------------------------------------------------------
// LayerNorm (f32 in, f16 out). One 256-thread block per row.
// ---------------------------------------------------------------------------
__global__ __launch_bounds__(256) void ln_kernel(const float* __restrict__ x,
                                                 const float* __restrict__ g,
                                                 const float* __restrict__ be,
                                                 half_t* __restrict__ y, int width) {
    int row = blockIdx.x;
    const float* xr = x + (size_t)row * width;
    float s = 0.f, s2 = 0.f;
    for (int i = threadIdx.x; i < width; i += 256) {
        float v = xr[i];
        s += v;
        s2 += v * v;
    }
#pragma unroll
    for (int msk = 16; msk; msk >>= 1) {
        s  += __shfl_xor(s,  msk, 32);
        s2 += __shfl_xor(s2, msk, 32);
    }
    __shared__ float rs[8], rs2[8];
    int wave = threadIdx.x >> 5;
    int lane = threadIdx.x & 31;
    if (lane == 0) { rs[wave] = s; rs2[wave] = s2; }
    __syncthreads();
    s = 0.f; s2 = 0.f;
#pragma unroll
    for (int w = 0; w < 8; ++w) { s += rs[w]; s2 += rs2[w]; }
    float mu   = s / (float)width;
    float var  = s2 / (float)width - mu * mu;
    float rstd = rsqrtf(var + 1e-5f);
    half_t* yr = y + (size_t)row * width;
    for (int i = threadIdx.x; i < width; i += 256)
        yr[i] = (half_t)((xr[i] - mu) * rstd * g[i] + be[i]);
}

// ---------------------------------------------------------------------------
// WMMA GEMM: C[M][N] = A[M][K] @ BT[N][K]^T + bias, f16 in / f32 acc.
// 256 threads = 8 waves; each wave owns 16 output rows, loops N in 16-col tiles.
// MODE 0: out = Qpad[b][h][row][d] f16, scaled by SCALE       (NROW = NQ)
// MODE 1: out = Kpad[b][h][row][d] f16                        (NROW = NKV)
// MODE 2: out = VT  [b][h][d][row] f16 (transposed store)     (NROW = NKV)
// MODE 3: out_f32 = resid + C  (final projection + residual)
// ---------------------------------------------------------------------------
template <int MODE>
__global__ __launch_bounds__(256) void gemm_wmma(const half_t* __restrict__ A,
                                                 const half_t* __restrict__ BT,
                                                 const float* __restrict__ bias,
                                                 int M, int N, int K, int NROW,
                                                 half_t* __restrict__ outH,
                                                 const float* __restrict__ resid,
                                                 float* __restrict__ outF) {
    const int lane = threadIdx.x & 31;
    const int wave = threadIdx.x >> 5;
    const int n    = lane & 15;   // C/D column, B row (in BT)
    const int hh   = lane >> 4;   // lane half
    const int row0 = (blockIdx.x * 8 + wave) * 16;
    if (row0 >= M) return;

    const half_t* Arow = A + (size_t)(row0 + n) * K;  // A row m = lane&15

    for (int n0 = 0; n0 < N; n0 += 16) {
        float bv = bias[n0 + n];
        v8f c = {bv, bv, bv, bv, bv, bv, bv, bv};
        const half_t* Brow = BT + (size_t)(n0 + n) * K;
        for (int k0 = 0; k0 < K; k0 += 32) {
            v16h a = load_fragA(Arow, k0, hh);
            v16h b = load_fragB(Brow, k0, hh);
            c = wmma_f16(a, b, c);
        }
#pragma unroll
        for (int r = 0; r < 8; ++r) {
            int R     = row0 + r + 8 * hh;
            float acc = c[r];
            if (MODE == 3) {
                size_t idx = (size_t)R * N + n0 + n;
                outF[idx]  = resid[idx] + acc;
            } else {
                int b_  = R / NROW;
                int rr  = R - b_ * NROW;
                int hed = n0 / HD;            // 16-col tiles never straddle heads
                int d   = n0 - hed * HD + n;  // (48 = 3*16)
                if (MODE == 0)
                    outH[((size_t)(b_ * NH + hed) * NROW + rr) * HDP + d] =
                        (half_t)(acc * SCALE);
                if (MODE == 1)
                    outH[((size_t)(b_ * NH + hed) * NROW + rr) * HDP + d] =
                        (half_t)acc;
                if (MODE == 2)
                    outH[((size_t)(b_ * NH + hed) * HDP + d) * NROW + rr] =
                        (half_t)acc;
            }
        }
    }
}

// ---------------------------------------------------------------------------
// Attention (transposed formulation): per wave, one 16-row q-tile over all
// 1024 keys, kv chunks of 64, online softmax.
//   S' = K . Q^T : A = K[kv][d] (natural), B = Q[q][d] (natural == BT)
//     -> D-tile rows = kv, cols = q  => softmax reduction is in-lane over
//        registers + ONE __shfl_xor(16); M/L/corr are per-lane scalars.
//   O' = V^T . P': A = VT[d][kv] (natural), B = P'[kv][q], BT = P[q][kv]
//     -> P packed to LDS as [q][kv] with one ds_store_b128 per 8 values,
//        read back as contiguous B fragments (reused across 3 d-tiles).
// LDS: double-buffered P tile, 2 x 8 waves x 16 x 64 f16 = 32 KB (of 320 KB).
// ---------------------------------------------------------------------------
__global__ __launch_bounds__(256) void attn_wmma(const half_t* __restrict__ Qb,
                                                 const half_t* __restrict__ Kb,
                                                 const half_t* __restrict__ VTb,
                                                 half_t* __restrict__ attnO) {
    __shared__ __align__(64) half_t pl[2][8][16][64];

    const int lane = threadIdx.x & 31;
    const int wave = threadIdx.x >> 5;
    const int n    = lane & 15;   // D column = q index; BT row
    const int hh   = lane >> 4;
    const int m    = lane & 15;   // A row (kv_local or d_local)

    int tile = blockIdx.x * 8 + wave;          // 8192 tiles total
    int qt   = tile & (QTILES - 1);
    int bh   = tile >> 9;                      // b*NH + head
    const half_t* Qp = Qb  + ((size_t)bh * NQ + qt * 16) * HDP;
    const half_t* Kp = Kb  + (size_t)bh * NKV * HDP;
    const half_t* Vp = VTb + (size_t)bh * HDP * NKV;

    // Q as B fragments (contiguous 32B loads), hoisted for the whole kernel
    v16h qb0 = load_fragB(Qp + n * HDP, 0, hh);
    v16h qb1 = load_fragB(Qp + n * HDP, 32, hh);

    float Mr = -1e30f, Lr = 0.f;               // per-lane (column q = n)
    v8f o0 = vzero8(), o1 = vzero8(), o2 = vzero8();   // O' d-tiles 0,16,32

    int buf = 0;
    for (int kv0 = 0; kv0 < NKV; kv0 += 64, buf ^= 1) {
        // ---- scores S'[kv][q]: 4 tiles of 16 kv rows, K-dim = 64 (padded) ----
        v8f s[4];
#pragma unroll
        for (int t = 0; t < 4; ++t) {
            const half_t* Krow = Kp + (size_t)(kv0 + t * 16 + m) * HDP;
            v8f st = vzero8();
            st = wmma_f16(load_fragA(Krow, 0, hh), qb0, st);
            st = wmma_f16(load_fragA(Krow, 32, hh), qb1, st);
            s[t] = st;
        }

        // ---- online softmax: reduce over kv (registers + one half-swap) ----
        float mx = -1e30f;
#pragma unroll
        for (int t = 0; t < 4; ++t)
#pragma unroll
            for (int r = 0; r < 8; ++r) mx = fmaxf(mx, s[t][r]);
        mx = fmaxf(mx, __shfl_xor(mx, 16, 32));
        float mn   = fmaxf(Mr, mx);
        float corr = exp2f((Mr - mn) * LOG2E);
        float rs   = 0.f;
#pragma unroll
        for (int t = 0; t < 4; ++t) {
            v8h ph;
#pragma unroll
            for (int r = 0; r < 8; ++r) {
                float p = exp2f((s[t][r] - mn) * LOG2E);
                rs += p;
                ph[r] = (half_t)p;
            }
            // P[q][kv]: 8 consecutive kv values -> one 16B LDS store
            *(v8h*)&pl[buf][wave][n][t * 16 + 8 * hh] = ph;
        }
        rs += __shfl_xor(rs, 16, 32);
        Lr = Lr * corr + rs;
        Mr = mn;
#pragma unroll
        for (int r = 0; r < 8; ++r) { o0[r] *= corr; o1[r] *= corr; o2[r] *= corr; }

        __syncthreads();

        // ---- O' += V^T . P' : P fragment reused across the 3 d-tiles ----
#pragma unroll
        for (int kk = 0; kk < 64; kk += 32) {
            v16h pf = *(const v16h*)&pl[buf][wave][n][kk + 16 * hh];
            o0 = wmma_f16(load_fragA(Vp + (size_t)(0  + m) * NKV, kv0 + kk, hh), pf, o0);
            o1 = wmma_f16(load_fragA(Vp + (size_t)(16 + m) * NKV, kv0 + kk, hh), pf, o1);
            o2 = wmma_f16(load_fragA(Vp + (size_t)(32 + m) * NKV, kv0 + kk, hh), pf, o2);
        }
    }

    // ---- epilogue: per-lane q row; 8 contiguous d per packed 16B store ----
    float inv = 1.f / Lr;
    half_t* rp = attnO + ((size_t)(bh >> 3) * NQ + qt * 16 + n) * CQ + (bh & 7) * HD;
    v8h w0, w1, w2;
#pragma unroll
    for (int r = 0; r < 8; ++r) {
        w0[r] = (half_t)(o0[r] * inv);
        w1[r] = (half_t)(o1[r] * inv);
        w2[r] = (half_t)(o2[r] * inv);
    }
    *(v8h*)(rp + 0  + 8 * hh) = w0;
    *(v8h*)(rp + 16 + 8 * hh) = w1;
    *(v8h*)(rp + 32 + 8 * hh) = w2;
}

// ---------------------------------------------------------------------------
// Host launcher
// ---------------------------------------------------------------------------
extern "C" void kernel_launch(void* const* d_in, const int* in_sizes, int n_in,
                              void* d_out, int out_size, void* d_ws, size_t ws_size,
                              hipStream_t stream) {
    (void)in_sizes; (void)n_in; (void)out_size; (void)ws_size;

    const float* qf    = (const float*)d_in[0];
    const float* kvf   = (const float*)d_in[1];
    const float* Wq    = (const float*)d_in[2];
    const float* bq    = (const float*)d_in[3];
    const float* Wk    = (const float*)d_in[4];
    const float* bk    = (const float*)d_in[5];
    const float* Wv    = (const float*)d_in[6];
    const float* bv    = (const float*)d_in[7];
    const float* Wo    = (const float*)d_in[8];
    const float* bo    = (const float*)d_in[9];
    const float* lnqg  = (const float*)d_in[10];
    const float* lnqb  = (const float*)d_in[11];
    const float* lnkvg = (const float*)d_in[12];
    const float* lnkvb = (const float*)d_in[13];
    float* out = (float*)d_out;

    // workspace carving (256B aligned)
    char* W = (char*)d_ws;
    size_t off = 0;
    auto take = [&](size_t bytes) -> char* {
        char* p = W + off;
        off += (bytes + 255) & ~(size_t)255;
        return p;
    };
    half_t* qn   = (half_t*)take((size_t)B_SZ * NQ * CQ * 2);        // 12.6 MB
    half_t* kvn  = (half_t*)take((size_t)B_SZ * NKV * CKV * 2);      //  3.1 MB
    half_t* WqT  = (half_t*)take((size_t)CQ * CQ * 2);
    half_t* WkT  = (half_t*)take((size_t)CQ * CKV * 2);
    half_t* WvT  = (half_t*)take((size_t)CQ * CKV * 2);
    half_t* WoT  = (half_t*)take((size_t)CQ * CQ * 2);
    size_t qb_bytes = (size_t)B_SZ * NH * NQ * HDP * 2;              // 16.8 MB
    size_t kb_bytes = (size_t)B_SZ * NH * NKV * HDP * 2;             //  2.1 MB
    half_t* Qb   = (half_t*)take(qb_bytes);
    half_t* Kb   = (half_t*)take(kb_bytes);
    half_t* VTb  = (half_t*)take(kb_bytes);
    half_t* attn = (half_t*)take((size_t)B_SZ * NQ * CQ * 2);        // 12.6 MB

    // zero padded head-dim regions (graph-capturable)
    hipMemsetAsync(Qb, 0, qb_bytes, stream);
    hipMemsetAsync(Kb, 0, kb_bytes, stream);
    hipMemsetAsync(VTb, 0, kb_bytes, stream);

    // weight convert + transpose to f16 [N][K]
    wconv_kernel<<<(CQ * CQ + 255) / 256, 256, 0, stream>>>(Wq, WqT, CQ, CQ);
    wconv_kernel<<<(CQ * CKV + 255) / 256, 256, 0, stream>>>(Wk, WkT, CKV, CQ);
    wconv_kernel<<<(CQ * CKV + 255) / 256, 256, 0, stream>>>(Wv, WvT, CKV, CQ);
    wconv_kernel<<<(CQ * CQ + 255) / 256, 256, 0, stream>>>(Wo, WoT, CQ, CQ);

    // layer norms -> f16
    ln_kernel<<<B_SZ * NQ, 256, 0, stream>>>(qf, lnqg, lnqb, qn, CQ);
    ln_kernel<<<B_SZ * NKV, 256, 0, stream>>>(kvf, lnkvg, lnkvb, kvn, CKV);

    // projections (WMMA): 128 rows per block
    gemm_wmma<0><<<(B_SZ * NQ) / 128, 256, 0, stream>>>(qn, WqT, bq,
        B_SZ * NQ, CQ, CQ, NQ, Qb, nullptr, nullptr);
    gemm_wmma<1><<<(B_SZ * NKV) / 128, 256, 0, stream>>>(kvn, WkT, bk,
        B_SZ * NKV, CQ, CKV, NKV, Kb, nullptr, nullptr);
    gemm_wmma<2><<<(B_SZ * NKV) / 128, 256, 0, stream>>>(kvn, WvT, bv,
        B_SZ * NKV, CQ, CKV, NKV, VTb, nullptr, nullptr);

    // attention: 8192 q-tiles, 8 waves per block
    attn_wmma<<<(B_SZ * NH * QTILES) / 8, 256, 0, stream>>>(Qb, Kb, VTb, attn);

    // output projection + bias + residual (f32 out)
    gemm_wmma<3><<<(B_SZ * NQ) / 128, 256, 0, stream>>>(attn, WoT, bo,
        B_SZ * NQ, CQ, CQ, NQ, nullptr, qf, out);
}
